// QLinearPerGrp_6262062317699
// MI455X (gfx1250) — compile-verified, matrix operand order checked by reading, not run on hardware
//
#include <hip/hip_runtime.h>

// ---------------------------------------------------------------------------
// QLinearPerGrp: y = x @ dequant(qweight).T + bias    (gfx1250 / MI455X)
//   x:       [1, 2048, 4096] bf16
//   qweight: [11008, 4096] int32 (values 0..15)
//   qzeros:  [11008, 32]  int32
//   scales:  [11008, 32]  float32
//   bias:    [11008]      float32
//   out:     [1, 2048, 11008] bf16
//
// Fused dequant (int4-in-int32 -> bf16 in LDS) + bf16 WMMA GEMM.
// Double-buffered LDS; x tile staged via async global->LDS (ASYNCcnt);
// qweight staged via VGPRs and dequantized (fma + fptrunc-to-bf16, which the
// backend can fuse into v_fma_mixlo/mixhi_bf16) while the current tile's
// WMMAs execute.
// ---------------------------------------------------------------------------

typedef __bf16 bf16_t;
typedef bf16_t v16bf __attribute__((ext_vector_type(16)));
typedef bf16_t v2bf  __attribute__((ext_vector_type(2)));
typedef float  v8f   __attribute__((ext_vector_type(8)));

#define BM 128
#define BN 128
#define BK 64
#define LDS_STRIDE 72   // BK + 8 elements of padding to spread LDS banks

#if defined(__has_builtin)
#if __has_builtin(__builtin_amdgcn_global_load_async_to_lds_b128)
#define HAVE_ASYNC_LDS 1
#endif
#if __has_builtin(__builtin_amdgcn_s_wait_asynccnt)
#define HAVE_WAIT_ASYNC 1
#endif
#endif

// GCC-style int vector matching the async builtin's parameter type in AS1/AS3.
typedef int vint4 __attribute__((vector_size(16)));
typedef __attribute__((address_space(1))) vint4 g_vint4;
typedef __attribute__((address_space(3))) vint4 s_vint4;

union FragV {
  uint4 u[2];
  v16bf v;
};

// Pack two f32 -> two bf16 in one dword using the compiler's native fptrunc
// (RNE; fusable with fma into v_fma_mixlo/mixhi_bf16 on CDNA5).
__device__ __forceinline__ unsigned int pack2bf(float lo, float hi) {
  v2bf r;
  r.x = (bf16_t)lo;
  r.y = (bf16_t)hi;
  return __builtin_bit_cast(unsigned int, r);
}

// Stage one thread's 64B slice of the x tile into LDS.
__device__ __forceinline__ void stage_x_tile(const bf16_t* gsrc, bf16_t* ldst) {
#if defined(HAVE_ASYNC_LDS)
  g_vint4* g = (g_vint4*)const_cast<bf16_t*>(gsrc);
  s_vint4* s = (s_vint4*)ldst;
  // IOFFSET applies to both the global and LDS addresses (ISA 15.18.3).
  __builtin_amdgcn_global_load_async_to_lds_b128(g, s, 0,  0);
  __builtin_amdgcn_global_load_async_to_lds_b128(g, s, 16, 0);
  __builtin_amdgcn_global_load_async_to_lds_b128(g, s, 32, 0);
  __builtin_amdgcn_global_load_async_to_lds_b128(g, s, 48, 0);
#else
  const uint4* src = (const uint4*)gsrc;
  uint4* dst = (uint4*)ldst;
  #pragma unroll
  for (int j = 0; j < 4; ++j) dst[j] = src[j];
#endif
}

__device__ __forceinline__ void wait_async_copies() {
#if defined(HAVE_ASYNC_LDS)
#if defined(HAVE_WAIT_ASYNC)
  __builtin_amdgcn_s_wait_asynccnt(0);
#else
  asm volatile("s_wait_asynccnt 0x0" ::: "memory");
#endif
#endif
}

// Dequantize one thread's 32 int32 weights -> 32 bf16 in LDS (4x b128 stores).
// (q - z) * sc == fma(float(q), sc, -z*sc)
__device__ __forceinline__ void dequant_store(bf16_t* ldst, const int4* q,
                                              float sc, float nzs) {
  uint4* dst = (uint4*)ldst;
  #pragma unroll
  for (int j = 0; j < 4; ++j) {
    int4 v0 = q[2 * j + 0];
    int4 v1 = q[2 * j + 1];
    uint4 o;
    o.x = pack2bf(__builtin_fmaf((float)v0.x, sc, nzs),
                  __builtin_fmaf((float)v0.y, sc, nzs));
    o.y = pack2bf(__builtin_fmaf((float)v0.z, sc, nzs),
                  __builtin_fmaf((float)v0.w, sc, nzs));
    o.z = pack2bf(__builtin_fmaf((float)v1.x, sc, nzs),
                  __builtin_fmaf((float)v1.y, sc, nzs));
    o.w = pack2bf(__builtin_fmaf((float)v1.z, sc, nzs),
                  __builtin_fmaf((float)v1.w, sc, nzs));
    dst[j] = o;
  }
}

__global__ __launch_bounds__(256)
void qlinear_wmma_bf16_kernel(const bf16_t* __restrict__ x,
                              const int*    __restrict__ qweight,
                              const int*    __restrict__ qzeros,
                              const float*  __restrict__ scales,
                              const float*  __restrict__ bias,
                              bf16_t* __restrict__ out,
                              int M, int N, int K, int G) {
  __shared__ bf16_t ldsA[2][BM * LDS_STRIDE];  // x tile      [m][k]
  __shared__ bf16_t ldsB[2][BN * LDS_STRIDE];  // bf16 W tile [n][k]

  const int tid  = threadIdx.x;
  const int lane = tid & 31;
  const int wave = tid >> 5;
  const int wm   = wave & 3;   // wave M position: wm*32
  const int wn   = wave >> 2;  // wave N position: wn*64

  const int n0 = blockIdx.x * BN;
  const int m0 = blockIdx.y * BM;

  // cooperative-loader assignment: 2 threads per row, 32 elements each
  const int lrow = tid >> 1;          // 0..127
  const int lk   = (tid & 1) * 32;    // 0 or 32

  const int GS = K / G;               // 128 (quant group size along K)
  const int nrow = n0 + lrow;         // qweight row this thread stages

  v8f acc[2][4];
  #pragma unroll
  for (int i = 0; i < 2; ++i)
    #pragma unroll
    for (int j = 0; j < 4; ++j)
      acc[i][j] = {};

  int4  qreg[8];
  float scN, nzsN;

  // ---- prologue: stage tile 0 into buffer 0 -------------------------------
  stage_x_tile(x + (size_t)(m0 + lrow) * K + lk,
               &ldsA[0][lrow * LDS_STRIDE + lk]);
  {
    const float sc = scales[(size_t)nrow * G];
    const int   z  = qzeros[(size_t)nrow * G];
    const int4* qsrc = (const int4*)(qweight + (size_t)nrow * K + lk);
    #pragma unroll
    for (int j = 0; j < 8; ++j) qreg[j] = qsrc[j];
    dequant_store(&ldsB[0][lrow * LDS_STRIDE + lk], qreg, sc, -(float)z * sc);
  }
  wait_async_copies();
  __syncthreads();

  // ---- main loop: compute buf, stage buf^1 --------------------------------
  for (int k0 = 0; k0 < K; k0 += BK) {
    const int  buf      = (k0 / BK) & 1;
    const bool has_next = (k0 + BK) < K;

    if (has_next) {
      const int kn = k0 + BK;
      stage_x_tile(x + (size_t)(m0 + lrow) * K + kn + lk,
                   &ldsA[buf ^ 1][lrow * LDS_STRIDE + lk]);
      const int g = kn / GS;
      scN = scales[(size_t)nrow * G + g];
      const int z = qzeros[(size_t)nrow * G + g];
      nzsN = -(float)z * scN;
      const int4* qsrc = (const int4*)(qweight + (size_t)nrow * K + kn + lk);
      #pragma unroll
      for (int j = 0; j < 8; ++j) qreg[j] = qsrc[j];
    }

    // ---- WMMA over this K-tile (two 16x16x32 steps) -----------------------
    const bf16_t* tileA = ldsA[buf];
    const bf16_t* tileB = ldsB[buf];
    #pragma unroll
    for (int kk = 0; kk < BK; kk += 32) {
      // ISA 16-bit A-fragment addressing: lanes 0-15 hold K {0..7,16..23},
      // lanes 16-31 hold K {8..15,24..31}; row/col = lane & 15.
      const int koff = kk + ((lane >> 4) << 3);
      FragV a[2], b[4];
      #pragma unroll
      for (int i = 0; i < 2; ++i) {
        const bf16_t* p = tileA + (wm * 32 + i * 16 + (lane & 15)) * LDS_STRIDE + koff;
        a[i].u[0] = *(const uint4*)p;
        a[i].u[1] = *(const uint4*)(p + 16);
      }
      #pragma unroll
      for (int j = 0; j < 4; ++j) {
        const bf16_t* p = tileB + (wn * 64 + j * 16 + (lane & 15)) * LDS_STRIDE + koff;
        b[j].u[0] = *(const uint4*)p;
        b[j].u[1] = *(const uint4*)(p + 16);
      }
      #pragma unroll
      for (int i = 0; i < 2; ++i)
        #pragma unroll
        for (int j = 0; j < 4; ++j)
          acc[i][j] = __builtin_amdgcn_wmma_f32_16x16x32_bf16(
              false, a[i].v, false, b[j].v, (short)0, acc[i][j], false, false);
    }

    if (has_next) {
      dequant_store(&ldsB[buf ^ 1][lrow * LDS_STRIDE + lk], qreg, scN, nzsN);
      wait_async_copies();   // next x tile resident before the barrier
    }
    __syncthreads();
  }

  // ---- epilogue: bias + convert to bf16 + store ---------------------------
  // C/D layout: VGPR r -> M = r + 8*(lane>=16), N = lane & 15.
  const int nlane = lane & 15;
  const int mhalf = (lane >> 4) << 3;
  #pragma unroll
  for (int j = 0; j < 4; ++j) {
    const int o = n0 + wn * 64 + j * 16 + nlane;
    const float bv = bias[o];
    #pragma unroll
    for (int i = 0; i < 2; ++i) {
      #pragma unroll
      for (int r = 0; r < 8; ++r) {
        const int m = m0 + wm * 32 + i * 16 + mhalf + r;
        out[(size_t)m * N + o] = (bf16_t)(acc[i][j][r] + bv);
      }
    }
  }
}

extern "C" void kernel_launch(void* const* d_in, const int* in_sizes, int n_in,
                              void* d_out, int out_size, void* d_ws, size_t ws_size,
                              hipStream_t stream) {
  const bf16_t* x       = (const bf16_t*)d_in[0];
  const int*    qweight = (const int*)d_in[1];
  const int*    qzeros  = (const int*)d_in[2];
  const float*  scales  = (const float*)d_in[3];
  const float*  bias    = (const float*)d_in[4];
  bf16_t*       out     = (bf16_t*)d_out;

  const int M = 2048, N = 11008, K = 4096, G = 32;
  dim3 grid(N / BN, M / BM);   // (86, 16)
  dim3 block(256);             // 8 wave32 waves
  qlinear_wmma_bf16_kernel<<<grid, block, 0, stream>>>(
      x, qweight, qzeros, scales, bias, out, M, N, K, G);
}
